// MultiHeadAttentionBlock_22428319220037
// MI455X (gfx1250) — matrix operand order repaired and checked by reading, not compiled, not run on hardware
//
#include <hip/hip_runtime.h>

// ---------------------------------------------------------------------------
// MultiHeadAttentionBlock for gfx1250 (MI455X, wave32, WMMA 16x16x32 f16)
// GN stats -> GN apply(+f16) -> w f32->f16 ->
// QKV WMMA GEMM   (double-buffered: async-LDS A, reg-staged transposed B) ->
// flash attention (S^T = K*Q^T, O^T = V^T*P^T, double-buffered async K / V) ->
// proj WMMA GEMM + residual
// ---------------------------------------------------------------------------

typedef _Float16 half_t;
typedef __attribute__((ext_vector_type(16))) _Float16 v16h;
typedef __attribute__((ext_vector_type(8)))  _Float16 v8h;
typedef __attribute__((ext_vector_type(8)))  float    v8f;

#define B_DIM 2
#define L_DIM 2048
#define C_DIM 1024
#define H_DIM 16
#define HD_DIM 64
#define G_DIM 32
#define CPG 32
#define M_DIM (B_DIM * L_DIM)   // 4096
#define N_QKV (3 * C_DIM)       // 3072

// ---- CDNA5 async global->LDS copy (16B per lane, ASYNCcnt tracked) ---------
static __device__ __forceinline__ void async_copy_b128(void* lds_dst,
                                                       const void* gbase,
                                                       unsigned byte_off) {
  unsigned lds_off = (unsigned)(size_t)lds_dst;   // addrspace(3) offset
  asm volatile("global_load_async_to_lds_b128 %0, %1, %2 offset:0"
               :: "v"(lds_off), "v"(byte_off), "s"((unsigned long long)gbase)
               : "memory");
}
static __device__ __forceinline__ void wait_async() {
  asm volatile("s_wait_asynccnt 0" ::: "memory");
}

// ---- WMMA fragment loaders (layouts per CDNA5 ISA 7.12.2, wave32) ----------
// A (16x32 f16): row = lane%16 ; K = (e<8?e:e+8) + (lane>=16?8:0)
static __device__ __forceinline__ v16h lds_a_frag(const half_t* t, int row, int ld,
                                                  int kk, int lane) {
  const half_t* p = t + (row + (lane & 15)) * ld + kk + ((lane >> 4) << 3);
  v16h a;
#pragma unroll
  for (int e = 0; e < 8; ++e) a[e] = p[e];
#pragma unroll
  for (int e = 0; e < 8; ++e) a[8 + e] = p[16 + e];
  return a;
}
// B (32x16 f16) from LDS stored as [col][k]: col = lane%16 ; K = (e<8?e:e+8) + (lane>=16?8:0)
static __device__ __forceinline__ v16h lds_b_frag(const half_t* t, int col, int ld,
                                                  int kk, int lane) {
  const half_t* p = t + (col + (lane & 15)) * ld + kk + ((lane >> 4) << 3);
  v16h b;
#pragma unroll
  for (int e = 0; e < 8; ++e) b[e] = p[e];
#pragma unroll
  for (int e = 0; e < 8; ++e) b[8 + e] = p[16 + e];
  return b;
}
static __device__ __forceinline__ v8f wmma_f16(v16h a, v16h b, v8f c) {
  return __builtin_amdgcn_wmma_f32_16x16x32_f16(false, a, false, b, (short)0, c,
                                                false, false);
}

// ---------------------------------------------------------------------------
// GroupNorm statistics: one block per (b,g); reduce over L x CPG = 65536 elems
// ---------------------------------------------------------------------------
__global__ __launch_bounds__(256) void gn_stats_kernel(const float* __restrict__ x,
                                                       float* __restrict__ stats) {
  const int bg = blockIdx.x;
  const int b = bg >> 5, g = bg & 31;
  const float* xb = x + (size_t)b * (L_DIM * C_DIM) + g * CPG;
  float s = 0.f, ss = 0.f;
  for (int i = threadIdx.x; i < L_DIM * CPG; i += 256) {
    float v = xb[(size_t)(i >> 5) * C_DIM + (i & 31)];
    s += v; ss += v * v;
  }
#pragma unroll
  for (int m = 16; m >= 1; m >>= 1) {
    s  += __shfl_xor(s, m, 32);
    ss += __shfl_xor(ss, m, 32);
  }
  __shared__ float rs[8], rss[8];
  const int lane = threadIdx.x & 31, wave = threadIdx.x >> 5;
  if (lane == 0) { rs[wave] = s; rss[wave] = ss; }
  __syncthreads();
  if (threadIdx.x == 0) {
    float S = 0.f, SS = 0.f;
#pragma unroll
    for (int w = 0; w < 8; ++w) { S += rs[w]; SS += rss[w]; }
    const float inv = 1.0f / (float)(L_DIM * CPG);
    float mean = S * inv;
    float var  = SS * inv - mean * mean;
    stats[bg * 2 + 0] = mean;
    stats[bg * 2 + 1] = rsqrtf(var + 1e-6f);
  }
}

__global__ __launch_bounds__(256) void gn_apply_kernel(const float* __restrict__ x,
                                                       const float* __restrict__ stats,
                                                       const float* __restrict__ scale,
                                                       const float* __restrict__ bias,
                                                       half_t* __restrict__ xn) {
  const size_t idx = (size_t)blockIdx.x * 256 + threadIdx.x;
  const int c = (int)(idx & (C_DIM - 1));
  const size_t row = idx >> 10;
  const int b = (int)(row >> 11);
  const int bg = b * G_DIM + (c >> 5);
  const float mean = stats[bg * 2 + 0];
  const float rstd = stats[bg * 2 + 1];
  xn[idx] = (half_t)((x[idx] - mean) * rstd * scale[c] + bias[c]);
}

__global__ __launch_bounds__(256) void f32_to_f16_kernel(const float* __restrict__ in,
                                                         half_t* __restrict__ out, int n) {
  int i = blockIdx.x * 256 + threadIdx.x;
  if (i < n) out[i] = (half_t)in[i];
}

// ---------------------------------------------------------------------------
// WMMA GEMM: C[M,N] = A[M,K]*B[K,N]. Block 128x128, 8 waves of 32x64.
// Double-buffered 2-stage pipeline:
//   top:    async A(i+1) -> As[nxt] ; global B(i+1) -> regs
//   middle: WMMA on As[cur], Bs[cur]
//   bottom: regs -> Bs[nxt] (transposed) ; s_wait_asynccnt 0 ; barrier
// MODE 0: qkv epilogue (bias, q*=1/8 on cols n%192<64, store f16)
// MODE 1: proj epilogue (bias, +residual, *1/sqrt2, store f32)
// ---------------------------------------------------------------------------
template <int MODE>
__global__ __launch_bounds__(256) void wmma_gemm_kernel(
    const half_t* __restrict__ A, const half_t* __restrict__ Bm,
    const float* __restrict__ bias, half_t* __restrict__ outH,
    const float* __restrict__ resid, float* __restrict__ outF,
    int M, int N, int K) {
  const int LDT = 40;                       // 80 B rows: 16B aligned
  __shared__ half_t As[2][128 * LDT];
  __shared__ half_t Bs[2][128 * LDT];

  const int tid = threadIdx.x, lane = tid & 31, wave = tid >> 5;
  const int wm = (wave >> 1) * 32, wn = (wave & 1) * 64;
  const int bm = blockIdx.y * 128, bn = blockIdx.x * 128;

  // fixed per-thread copy coordinates
  const int ar = tid >> 2, akg = (tid & 3) * 8;       // A chunks: rows ar+64g
  const int bkr = tid >> 4, bng = (tid & 15) * 8;     // B chunks: k rows bkr+16g

  v8f acc[2][4] = {};
  v8h breg[2];
  const int nsteps = K / 32;

  // ---- prologue: stage tile 0 ----
#pragma unroll
  for (int g = 0; g < 4; ++g)
    async_copy_b128(&As[0][(ar + g * 64) * LDT + akg], A,
                    (unsigned)(((bm + ar + g * 64) * K + akg) * 2));
#pragma unroll
  for (int g = 0; g < 2; ++g)
    breg[g] = *(const v8h*)(Bm + (size_t)(bkr + g * 16) * N + bn + bng);
#pragma unroll
  for (int g = 0; g < 2; ++g)
#pragma unroll
    for (int e = 0; e < 8; ++e)
      Bs[0][(bng + e) * LDT + (bkr + g * 16)] = breg[g][e];
  wait_async();
  __syncthreads();

  for (int i = 0; i < nsteps; ++i) {
    const int cur = i & 1, nxt = cur ^ 1;
    const int k1 = (i + 1) * 32;
    if (i + 1 < nsteps) {
#pragma unroll
      for (int g = 0; g < 4; ++g)
        async_copy_b128(&As[nxt][(ar + g * 64) * LDT + akg], A,
                        (unsigned)(((bm + ar + g * 64) * K + k1 + akg) * 2));
#pragma unroll
      for (int g = 0; g < 2; ++g)
        breg[g] = *(const v8h*)(Bm + (size_t)(k1 + bkr + g * 16) * N + bn + bng);
    }

    const v16h a0 = lds_a_frag(As[cur], wm + 0,  LDT, 0, lane);
    const v16h a1 = lds_a_frag(As[cur], wm + 16, LDT, 0, lane);
#pragma unroll
    for (int ni = 0; ni < 4; ++ni) {
      const v16h bf = lds_b_frag(Bs[cur], wn + ni * 16, LDT, 0, lane);
      acc[0][ni] = wmma_f16(a0, bf, acc[0][ni]);
      acc[1][ni] = wmma_f16(a1, bf, acc[1][ni]);
    }

    if (i + 1 < nsteps) {
#pragma unroll
      for (int g = 0; g < 2; ++g)
#pragma unroll
        for (int e = 0; e < 8; ++e)
          Bs[nxt][(bng + e) * LDT + (bkr + g * 16)] = breg[g][e];
    }
    wait_async();
    __syncthreads();
  }

#pragma unroll
  for (int mi = 0; mi < 2; ++mi)
#pragma unroll
    for (int ni = 0; ni < 4; ++ni) {
      const v8f c = acc[mi][ni];
      const int col = bn + wn + ni * 16 + (lane & 15);
      const int rbase = bm + wm + mi * 16 + ((lane >> 4) << 3);
#pragma unroll
      for (int v = 0; v < 8; ++v) {
        const int row = rbase + v;
        float val = c[v] + bias[col];
        if (MODE == 0) {
          if ((col % 192) < 64) val *= 0.125f;     // q * 1/sqrt(hd)
          outH[(size_t)row * N + col] = (half_t)val;
        } else {
          outF[(size_t)row * N + col] =
              (resid[(size_t)row * N + col] + val) * 0.70710678118654752f;
        }
      }
    }
}

// ---------------------------------------------------------------------------
// Flash attention, transposed-logits form, double-buffered key tiles.
// grid.y = b*H, grid.x = 128-query tile. Wave w owns queries [16w,16w+16):
//   S^T(64key x 16q) = K(64x64) * Q^T(64x16)   -> per-lane softmax
//   O^T(64hd x 16q) += V^T(64x64) * P^T(64x16)
// ---------------------------------------------------------------------------
__global__ __launch_bounds__(256) void flash_attn_kernel(
    const half_t* __restrict__ qkv, half_t* __restrict__ outH) {
  const int LDT = 72;                       // 144 B rows: 16B aligned
  __shared__ half_t Qs[128 * LDT];          // [query][hd]   (async, resident)
  __shared__ half_t Ks[2][64 * LDT];        // [key][hd]     (async)
  __shared__ half_t Vt[2][64 * LDT];        // [hd][key]     (reg-staged transpose)
  __shared__ half_t Ps[8][16 * LDT];        // per-wave P^T: [query][key]

  const int bh = blockIdx.y;
  const int b = bh / H_DIM, h = bh % H_DIM;
  const int tid = threadIdx.x, lane = tid & 31, wave = tid >> 5;
  const size_t rowbase = (size_t)b * L_DIM;
  const int qcol = h * 192, kcol = qcol + 64, vcol = qcol + 128;
  const int q0 = blockIdx.x * 128;

  const int kr = tid >> 3, kcg = (tid & 7) * 8;       // K/V chunks: rows kr+32g

  // ---- prologue: Q tile (resident) + key tile 0 ----
#pragma unroll
  for (int g = 0; g < 4; ++g) {
    const int r = (tid >> 3) + g * 32;                // 0..127
    async_copy_b128(Qs + r * LDT + kcg, qkv,
                    (unsigned)(((unsigned)(rowbase + q0 + r) * N_QKV + qcol + kcg) * 2));
  }
  v8h vreg[2];
#pragma unroll
  for (int g = 0; g < 2; ++g) {
    const int r = kr + g * 32;
    async_copy_b128(&Ks[0][r * LDT + kcg], qkv,
                    (unsigned)(((unsigned)(rowbase + r) * N_QKV + kcol + kcg) * 2));
    vreg[g] = *(const v8h*)(qkv + (rowbase + r) * N_QKV + vcol + kcg);
  }
#pragma unroll
  for (int g = 0; g < 2; ++g)
#pragma unroll
    for (int e = 0; e < 8; ++e)
      Vt[0][(kcg + e) * LDT + (kr + g * 32)] = vreg[g][e];
  wait_async();
  __syncthreads();

  float m = -1e30f, l = 0.f;
  v8f o[4] = {};
  half_t* ps = &Ps[wave][0];
  const int nkt = L_DIM / 64;

  for (int it = 0; it < nkt; ++it) {
    const int cur = it & 1, nxt = cur ^ 1;
    const int kt1 = (it + 1) * 64;
    if (it + 1 < nkt) {
#pragma unroll
      for (int g = 0; g < 2; ++g) {
        const int r = kr + g * 32;
        async_copy_b128(&Ks[nxt][r * LDT + kcg], qkv,
                        (unsigned)(((unsigned)(rowbase + kt1 + r) * N_QKV + kcol + kcg) * 2));
        vreg[g] = *(const v8h*)(qkv + (rowbase + kt1 + r) * N_QKV + vcol + kcg);
      }
    }

    // S^T strip: Q fragments shared across the 4 key tiles
    const v16h qf0 = lds_b_frag(Qs, wave * 16, LDT, 0,  lane);
    const v16h qf1 = lds_b_frag(Qs, wave * 16, LDT, 32, lane);
    v8f st[4];
#pragma unroll
    for (int ki = 0; ki < 4; ++ki) {
      v8f a = {};
      a = wmma_f16(lds_a_frag(Ks[cur], ki * 16, LDT, 0,  lane), qf0, a);
      a = wmma_f16(lds_a_frag(Ks[cur], ki * 16, LDT, 32, lane), qf1, a);
      st[ki] = a;
    }

    // Per-lane softmax (query = lane%16; merge key halves with one shuffle)
    float mx = st[0][0];
#pragma unroll
    for (int ki = 0; ki < 4; ++ki)
#pragma unroll
      for (int v = 0; v < 8; ++v) mx = fmaxf(mx, st[ki][v]);
    mx = fmaxf(mx, __shfl_xor(mx, 16, 32));
    const float mnew = fmaxf(m, mx);
    const float sc = __expf(m - mnew);
    float rsum = 0.f;
#pragma unroll
    for (int ki = 0; ki < 4; ++ki)
#pragma unroll
      for (int v = 0; v < 8; ++v) {
        const float p = __expf(st[ki][v] - mnew);
        st[ki][v] = p;
        rsum += p;
      }
    rsum += __shfl_xor(rsum, 16, 32);
    l = l * sc + rsum;
    m = mnew;

    // P^T -> wave-private LDS [query][key]
#pragma unroll
    for (int ki = 0; ki < 4; ++ki)
#pragma unroll
      for (int v = 0; v < 8; ++v)
        ps[(lane & 15) * LDT + ki * 16 + ((lane >> 4) << 3) + v] = (half_t)st[ki][v];
    asm volatile("s_wait_dscnt 0" ::: "memory");   // wave-private fence

    // O^T rescale + V^T(64hd x 64key) * P^T(64key x 16q)
#pragma unroll
    for (int mi = 0; mi < 4; ++mi)
#pragma unroll
      for (int v = 0; v < 8; ++v) o[mi][v] *= sc;
    const v16h pf0 = lds_b_frag(ps, 0, LDT, 0,  lane);
    const v16h pf1 = lds_b_frag(ps, 0, LDT, 32, lane);
#pragma unroll
    for (int mi = 0; mi < 4; ++mi) {
      o[mi] = wmma_f16(lds_a_frag(Vt[cur], mi * 16, LDT, 0,  lane), pf0, o[mi]);
      o[mi] = wmma_f16(lds_a_frag(Vt[cur], mi * 16, LDT, 32, lane), pf1, o[mi]);
    }

    if (it + 1 < nkt) {
#pragma unroll
      for (int g = 0; g < 2; ++g)
#pragma unroll
        for (int e = 0; e < 8; ++e)
          Vt[nxt][(kcg + e) * LDT + (kr + g * 32)] = vreg[g][e];
    }
    wait_async();
    __syncthreads();
  }

  // Output: query = lane%16, hd = mi*16 + 8*(lane>=16) + v -> 16B stores
  const float linv = 1.0f / l;
  const size_t orow = (rowbase + q0 + wave * 16 + (lane & 15)) * C_DIM + h * HD_DIM;
#pragma unroll
  for (int mi = 0; mi < 4; ++mi) {
    v8h t;
#pragma unroll
    for (int v = 0; v < 8; ++v) t[v] = (half_t)(o[mi][v] * linv);
    *(v8h*)(outH + orow + mi * 16 + ((lane >> 4) << 3)) = t;
  }
}

// ---------------------------------------------------------------------------
extern "C" void kernel_launch(void* const* d_in, const int* in_sizes, int n_in,
                              void* d_out, int out_size, void* d_ws, size_t ws_size,
                              hipStream_t stream) {
  const float* x        = (const float*)d_in[0];
  const float* gn_scale = (const float*)d_in[1];
  const float* gn_bias  = (const float*)d_in[2];
  const float* qkv_w    = (const float*)d_in[3];
  const float* qkv_b    = (const float*)d_in[4];
  const float* proj_w   = (const float*)d_in[5];
  const float* proj_b   = (const float*)d_in[6];
  float* out = (float*)d_out;

  char* w = (char*)d_ws;
  float*  stats = (float*)w;                                     // 512 B
  half_t* xn    = (half_t*)(w + 1024);                           // 4096x1024
  half_t* qkvw  = xn   + (size_t)M_DIM * C_DIM;                  // 1024x3072
  half_t* projw = qkvw + (size_t)C_DIM * N_QKV;                  // 1024x1024
  half_t* qkvo  = projw + (size_t)C_DIM * C_DIM;                 // 4096x3072
  half_t* hbuf  = qkvo + (size_t)M_DIM * N_QKV;                  // 4096x1024

  gn_stats_kernel<<<B_DIM * G_DIM, 256, 0, stream>>>(x, stats);
  gn_apply_kernel<<<(M_DIM * C_DIM) / 256, 256, 0, stream>>>(x, stats, gn_scale,
                                                             gn_bias, xn);
  f32_to_f16_kernel<<<(C_DIM * N_QKV + 255) / 256, 256, 0, stream>>>(qkv_w, qkvw,
                                                                     C_DIM * N_QKV);
  f32_to_f16_kernel<<<(C_DIM * C_DIM + 255) / 256, 256, 0, stream>>>(proj_w, projw,
                                                                     C_DIM * C_DIM);
  wmma_gemm_kernel<0><<<dim3(N_QKV / 128, M_DIM / 128), 256, 0, stream>>>(
      xn, qkvw, qkv_b, qkvo, nullptr, nullptr, M_DIM, N_QKV, C_DIM);
  flash_attn_kernel<<<dim3(L_DIM / 128, B_DIM * H_DIM), 256, 0, stream>>>(qkvo, hbuf);
  wmma_gemm_kernel<1><<<dim3(C_DIM / 128, M_DIM / 128), 256, 0, stream>>>(
      hbuf, projw, proj_b, nullptr, x, out, M_DIM, C_DIM, C_DIM);
}
